// Qwen3_5GatedDeltaNet_61332132987711
// MI455X (gfx1250) — matrix operand order repaired and checked by reading, not compiled, not run on hardware
//
#include <hip/hip_runtime.h>

typedef __bf16 bhalf_t;
typedef __attribute__((ext_vector_type(16))) __bf16 v16bf;
typedef __attribute__((ext_vector_type(8)))  __bf16 v8bf;
typedef __attribute__((ext_vector_type(8)))  float  v8f;
typedef unsigned int       u32;
typedef unsigned long long u64;
typedef u32 u32x4 __attribute__((ext_vector_type(4)));
typedef u32 u32x8 __attribute__((ext_vector_type(8)));

// ---------------------------------------------------------------------------
// TDM: 2-D tile load, global -> LDS.  D# layout per CDNA5 ISA
// group0: count=1, lds_addr, global_addr(57b), type=2
// group1: data_size=2 (4-byte units), tensor_dim0/1, tile_dim0/1, dim0 stride
// Issued once per wave (EXEC ignored); tracked on TENSORcnt.
// ---------------------------------------------------------------------------
__device__ __forceinline__ void tdm_load_2d(u32 lds_addr, const void* gaddr,
                                            u32 tensor_d0, u32 tensor_d1,
                                            u32 tile_d0, u32 tile_d1,
                                            u32 stride_d0)
{
    u64 ga = (u64)gaddr;
    u32x4 g0;
    g0[0] = 1u;                                   // count=1, user mode
    g0[1] = lds_addr;                             // LDS byte address
    g0[2] = (u32)ga;                              // global_addr[31:0]
    g0[3] = (u32)(ga >> 32) | 0x80000000u;        // addr[56:32] | type=2<<30
    u32x8 g1;
    g1[0] = 0x00020000u;                          // data_size=2 -> 4B units
    g1[1] = (tensor_d0 & 0xFFFFu) << 16;          // tensor_dim0[15:0]
    g1[2] = ((tensor_d0 >> 16) & 0xFFFFu) | ((tensor_d1 & 0xFFFFu) << 16);
    g1[3] = ((tensor_d1 >> 16) & 0xFFFFu) | (tile_d0 << 16);
    g1[4] = tile_d1;                              // tile_dim1 | tile_dim2=0
    g1[5] = stride_d0;                            // dim0 stride (4B units)
    g1[6] = 0u;
    g1[7] = 0u;
    u32x4 gz = {0u, 0u, 0u, 0u};
    asm volatile("tensor_load_to_lds %0, %1, %2, %3"
                 :: "s"(g0), "s"(g1), "s"(gz), "s"(gz)
                 : "memory");
}

// ---------------------------------------------------------------------------
// f32 -> bf16 cast (row major preserved)
// ---------------------------------------------------------------------------
__global__ void k_cast_bf16(const float* __restrict__ in, bhalf_t* __restrict__ out,
                            size_t n) {
    size_t i = ((size_t)blockIdx.x * blockDim.x + threadIdx.x) * 4;
    if (i < n) {
#pragma unroll
        for (int j = 0; j < 4; ++j) out[i + j] = (bhalf_t)in[i + j];
    }
}

// ---------------------------------------------------------------------------
// f32 [K][N] -> bf16 transposed [N][K] (32x32 LDS tiles, K,N % 32 == 0)
// ---------------------------------------------------------------------------
__global__ __launch_bounds__(256) void k_castT_bf16(
    const float* __restrict__ in, bhalf_t* __restrict__ out, int K, int N)
{
    __shared__ float tile[32][33];
    int kb = blockIdx.y * 32, nb = blockIdx.x * 32;
    int tx = threadIdx.x & 31, ty = threadIdx.x >> 5;   // 32 x 8
#pragma unroll
    for (int i = 0; i < 32; i += 8)
        tile[ty + i][tx] = in[(size_t)(kb + ty + i) * N + nb + tx];
    __syncthreads();
#pragma unroll
    for (int i = 0; i < 32; i += 8)
        out[(size_t)(nb + ty + i) * K + kb + tx] = (bhalf_t)tile[tx][ty + i];
}

// ---------------------------------------------------------------------------
// bf16 GEMM, TDM-staged, double buffered, v_wmma_f32_16x16x32_bf16.
// C[M,N] f32 = A[M,K] bf16 (row major) @ Bt[N,K] bf16 (pre-transposed).
// Workgroup tile 128x64, K-step 32; 8 waves as 4x2 of 32x32 wave tiles.
// M%128==0, N%64==0, K%32==0.
// ---------------------------------------------------------------------------
#define TM 128
#define TN 64
#define TK 32

__global__ __launch_bounds__(256) void k_gemm_bf16(
    const bhalf_t* __restrict__ A, const bhalf_t* __restrict__ Bt,
    float* __restrict__ C, int M, int N, int K)
{
    __shared__ bhalf_t As[2][TM][TK];   // 2 x 8 KB
    __shared__ bhalf_t Bs[2][TN][TK];   // 2 x 4 KB

    const int tid  = threadIdx.x;
    const int lane = tid & 31;
    const int wave = tid >> 5;
    const int wm   = wave & 3;
    const int wn   = wave >> 2;
    const int m0   = blockIdx.y * TM;
    const int n0   = blockIdx.x * TN;
    const int nsteps = K / TK;
    const u32 kdw  = (u32)(K >> 1);     // K in 4-byte units

    v8f acc[2][2] = {};

    auto issue = [&](int step, int buf) {
        int k0 = step * TK;
        tdm_load_2d((u32)(size_t)&As[buf][0][0], A + (size_t)m0 * K + k0,
                    kdw, (u32)M, TK / 2, TM, kdw);
        tdm_load_2d((u32)(size_t)&Bs[buf][0][0], Bt + (size_t)n0 * K + k0,
                    kdw, (u32)N, TK / 2, TN, kdw);
    };

    if (wave == 0) issue(0, 0);

    for (int step = 0; step < nsteps; ++step) {
        const int buf = step & 1;
        if (wave == 0) {
            if (step + 1 < nsteps) {
                issue(step + 1, buf ^ 1);
                __builtin_amdgcn_s_wait_tensorcnt(2);  // retire step's pair
            } else {
                __builtin_amdgcn_s_wait_tensorcnt(0);  // drain
            }
        }
        __syncthreads();

        // A frag: lane L -> row L%16; K = g*8..g*8+7 and 16+g*8.., g=L/16
        v16bf afr[2], bfr[2];
#pragma unroll
        for (int mi = 0; mi < 2; ++mi) {
            int row = wm * 32 + mi * 16 + (lane & 15);
            int kb  = (lane >> 4) * 8;
            v8bf lo = *(const v8bf*)&As[buf][row][kb];
            v8bf hi = *(const v8bf*)&As[buf][row][16 + kb];
#pragma unroll
            for (int i = 0; i < 8; ++i) { afr[mi][i] = lo[i]; afr[mi][8 + i] = hi[i]; }
        }
        // B frag: lane L -> col L%16; K = g*8..g*8+7 and 16+g*8.., g=L/16
#pragma unroll
        for (int ni = 0; ni < 2; ++ni) {
            int col = wn * 32 + ni * 16 + (lane & 15);
            int kb  = (lane >> 4) * 8;
            v8bf lo = *(const v8bf*)&Bs[buf][col][kb];
            v8bf hi = *(const v8bf*)&Bs[buf][col][16 + kb];
#pragma unroll
            for (int i = 0; i < 8; ++i) { bfr[ni][i] = lo[i]; bfr[ni][8 + i] = hi[i]; }
        }

#pragma unroll
        for (int mi = 0; mi < 2; ++mi)
#pragma unroll
            for (int ni = 0; ni < 2; ++ni)
                acc[mi][ni] = __builtin_amdgcn_wmma_f32_16x16x32_bf16(
                    false, afr[mi], false, bfr[ni], (short)0, acc[mi][ni],
                    false, false);
        __syncthreads();   // protect buf before TDM overwrites it next+1 step
    }

    // Epilogue: VGPR r, lane L -> M = r + 8*(L/16), N = L%16
#pragma unroll
    for (int mi = 0; mi < 2; ++mi)
#pragma unroll
        for (int ni = 0; ni < 2; ++ni) {
            int col   = n0 + wn * 32 + ni * 16 + (lane & 15);
            int rbase = m0 + wm * 32 + mi * 16 + ((lane >> 4) << 3);
            float* cp = C + (size_t)rbase * N + col;
#pragma unroll
            for (int r = 0; r < 8; ++r) cp[(size_t)r * N] = acc[mi][ni][r];
        }
}

// ---------------------------------------------------------------------------
// beta = sigmoid(X @ W_b), g = -exp(A_log) * softplus(X @ W_a + dt_bias)
// ---------------------------------------------------------------------------
__global__ __launch_bounds__(64) void k_proj_gate(
    const float* __restrict__ hs, const float* __restrict__ Wb,
    const float* __restrict__ Wa, const float* __restrict__ dtb,
    const float* __restrict__ Alog, float* __restrict__ beta,
    float* __restrict__ g, int D)
{
    int row = blockIdx.x;
    int j   = threadIdx.x;
    int col = j & 31;
    const float* x = hs + (size_t)row * D;
    const float* W = (j < 32) ? Wb : Wa;
    float acc = 0.f;
    for (int i = 0; i < D; ++i) acc = fmaf(x[i], W[(size_t)i * 32 + col], acc);
    if (j < 32) {
        beta[(size_t)row * 32 + col] = 1.f / (1.f + __expf(-acc));
    } else {
        float xg = acc + dtb[col];
        float sp = (xg > 20.f) ? xg : log1pf(__expf(xg));
        g[(size_t)row * 32 + col] = -__expf(Alog[col]) * sp;
    }
}

// ---------------------------------------------------------------------------
// Causal depthwise conv (K=4) + silu
// ---------------------------------------------------------------------------
__global__ __launch_bounds__(256) void k_conv_silu(
    const float* __restrict__ mixed, const float* __restrict__ cw,
    float* __restrict__ out, int T, int Cd)
{
    size_t idx = (size_t)blockIdx.x * blockDim.x + threadIdx.x;
    int c = (int)(idx % Cd);
    size_t bt = idx / Cd;
    int t = (int)(bt % T);
    const float* w = cw + (size_t)c * 4;
    float acc = 0.f;
#pragma unroll
    for (int i = 0; i < 4; ++i) {
        int tt = t - 3 + i;
        if (tt >= 0) acc = fmaf(mixed[idx + (size_t)(i - 3) * Cd], w[i], acc);
    }
    out[idx] = acc / (1.f + __expf(-acc));
}

// ---------------------------------------------------------------------------
// In-place l2norm of q (scaled DK^-0.5) and k; one wave per 128-group.
// ---------------------------------------------------------------------------
__global__ __launch_bounds__(256) void k_qknorm(float* __restrict__ conv)
{
    int lane = threadIdx.x & 31;
    int wv   = threadIdx.x >> 5;
    size_t gidx = (size_t)blockIdx.x * 8 + wv;
    int which = (int)(gidx & 31);
    size_t bt = gidx >> 5;
    float* p = conv + bt * 8192 +
               (which < 16 ? (size_t)which * 128 : 2048 + (size_t)(which - 16) * 128);
    float a0 = p[lane], a1 = p[lane + 32], a2 = p[lane + 64], a3 = p[lane + 96];
    float ss = a0 * a0 + a1 * a1 + a2 * a2 + a3 * a3;
#pragma unroll
    for (int m = 16; m >= 1; m >>= 1) ss += __shfl_xor(ss, m, 32);
    float scale = rsqrtf(ss + 1e-6f);
    if (which < 16) scale *= 0.08838834764831845f;  // 128^-0.5
    p[lane] = a0 * scale; p[lane + 32] = a1 * scale;
    p[lane + 64] = a2 * scale; p[lane + 96] = a3 * scale;
}

// ---------------------------------------------------------------------------
// Gated delta rule. One block per (b, value-head); lane pair per v-column:
// thread = (v, row-half); 64 f32 state regs per thread; pair-combine via shfl.
// ---------------------------------------------------------------------------
__global__ __launch_bounds__(256) void k_delta(
    const float* __restrict__ conv, const float* __restrict__ g,
    const float* __restrict__ beta, float* __restrict__ o, int T)
{
    const int b   = blockIdx.x >> 5;
    const int h   = blockIdx.x & 31;
    const int hk  = h >> 1;            // rep = NV/NK = 2
    const int tid = threadIdx.x;       // 0..255
    const int v   = tid >> 1;          // v column 0..127
    const int rh  = tid & 1;           // row half
    const int rbase = rh * 64;
    __shared__ float kq[256];

    float s[64];
#pragma unroll
    for (int j = 0; j < 64; ++j) s[j] = 0.f;

    const float* base = conv + (size_t)b * T * 8192;
    const float* gb   = g    + (size_t)b * T * 32 + h;
    const float* bb   = beta + (size_t)b * T * 32 + h;

    for (int t = 0; t < T; ++t) {
        const float* row = base + (size_t)t * 8192;
        if (tid < 128) {
            kq[tid]       = row[2048 + hk * 128 + tid];  // k
            kq[128 + tid] = row[hk * 128 + tid];         // q
        }
        __syncthreads();
        float vt = row[4096 + h * 128 + v];
        float eg = __expf(gb[(size_t)t * 32]);
        float bt = bb[(size_t)t * 32];

        float kd = 0.f;
#pragma unroll
        for (int j = 0; j < 64; ++j) { s[j] *= eg; kd = fmaf(kq[rbase + j], s[j], kd); }
        kd += __shfl_xor(kd, 1, 32);
        float coef = bt * (vt - kd);
        float od = 0.f;
#pragma unroll
        for (int j = 0; j < 64; ++j) {
            s[j] = fmaf(coef, kq[rbase + j], s[j]);
            od   = fmaf(kq[128 + rbase + j], s[j], od);
        }
        od += __shfl_xor(od, 1, 32);
        if (rh == 0) o[((size_t)(b * T + t) * 32 + h) * 128 + v] = od;
        __syncthreads();
    }
}

// ---------------------------------------------------------------------------
// Fused gated RMSNorm -> bf16
// ---------------------------------------------------------------------------
__global__ __launch_bounds__(256) void k_gatednorm(
    const float* __restrict__ o, const float* __restrict__ z,
    const float* __restrict__ nw, bhalf_t* __restrict__ out)
{
    int lane = threadIdx.x & 31;
    int wv   = threadIdx.x >> 5;
    size_t gidx = (size_t)blockIdx.x * 8 + wv;
    const float* op = o + gidx * 128;
    const float* zp = z + gidx * 128;
    bhalf_t* wp = out + gidx * 128;
    float a[4];
    float ss = 0.f;
#pragma unroll
    for (int i = 0; i < 4; ++i) { a[i] = op[lane + 32 * i]; ss = fmaf(a[i], a[i], ss); }
#pragma unroll
    for (int m = 16; m >= 1; m >>= 1) ss += __shfl_xor(ss, m, 32);
    float r = rsqrtf(ss * (1.f / 128.f) + 1e-6f);
#pragma unroll
    for (int i = 0; i < 4; ++i) {
        float zz = zp[lane + 32 * i];
        float sz = zz / (1.f + __expf(-zz));
        wp[lane + 32 * i] = (bhalf_t)(a[i] * r * nw[lane + 32 * i] * sz);
    }
}

// ---------------------------------------------------------------------------
extern "C" void kernel_launch(void* const* d_in, const int* in_sizes, int n_in,
                              void* d_out, int out_size, void* d_ws, size_t ws_size,
                              hipStream_t stream) {
    const float* hs    = (const float*)d_in[0];
    const float* Wqkv  = (const float*)d_in[1];
    const float* Wz    = (const float*)d_in[2];
    const float* Wb    = (const float*)d_in[3];
    const float* Wa    = (const float*)d_in[4];
    const float* convw = (const float*)d_in[5];
    const float* dtb   = (const float*)d_in[6];
    const float* Alog  = (const float*)d_in[7];
    const float* normw = (const float*)d_in[8];
    const float* Wout  = (const float*)d_in[9];
    float* out = (float*)d_out;

    const int Bb = 2, T = 4096, D = 2048, CD = 8192, VD = 4096;
    const size_t BT = (size_t)Bb * T;  // 8192

    char* ws = (char*)d_ws;
    bhalf_t* hsB   = (bhalf_t*)(ws);                                  // 33.5 MB
    bhalf_t* wbuf  = (bhalf_t*)(ws + 33554432ull);                    // 33.5 MB (reused, transposed weights)
    float*   mixed = (float*)(ws + 67108864ull);                      // 268 MB
    float*   convb = (float*)(ws + 67108864ull + 268435456ull);       // 268 MB
    float*   zbuf  = (float*)(ws + 67108864ull + 2 * 268435456ull);   // 134 MB
    float*   gbuf  = (float*)(ws + 67108864ull + 2 * 268435456ull + 134217728ull);
    float*   betab = gbuf + 262144;
    float*   obuf  = mixed;                                           // reuse
    bhalf_t* ognB  = (bhalf_t*)((char*)mixed + 134217728ull);         // reuse tail

    // hidden -> bf16 (row major)
    {
        size_t n = BT * D;
        k_cast_bf16<<<(int)((n + 1023) / 1024), 256, 0, stream>>>(hs, hsB, n);
    }

    // mixed = X @ W_qkv
    k_castT_bf16<<<dim3(CD / 32, D / 32), 256, 0, stream>>>(Wqkv, wbuf, D, CD);
    k_gemm_bf16<<<dim3(CD / TN, (int)BT / TM), 256, 0, stream>>>(
        hsB, wbuf, mixed, (int)BT, CD, D);

    // z = X @ W_z
    k_castT_bf16<<<dim3(VD / 32, D / 32), 256, 0, stream>>>(Wz, wbuf, D, VD);
    k_gemm_bf16<<<dim3(VD / TN, (int)BT / TM), 256, 0, stream>>>(
        hsB, wbuf, zbuf, (int)BT, VD, D);

    k_proj_gate<<<(int)BT, 64, 0, stream>>>(hs, Wb, Wa, dtb, Alog, betab, gbuf, D);

    k_conv_silu<<<(int)(BT * CD / 256), 256, 0, stream>>>(mixed, convw, convb, T, CD);

    k_qknorm<<<(int)(BT * 32 / 8), 256, 0, stream>>>(convb);

    k_delta<<<Bb * 32, 256, 0, stream>>>(convb, gbuf, betab, obuf, T);

    k_gatednorm<<<(int)(BT * 32 / 8), 256, 0, stream>>>(obuf, zbuf, normw, ognB);

    // out = ognorm @ W_out
    k_castT_bf16<<<dim3(D / 32, VD / 32), 256, 0, stream>>>(Wout, wbuf, VD, D);
    k_gemm_bf16<<<dim3(D / TN, (int)BT / TM), 256, 0, stream>>>(
        ognB, wbuf, out, (int)BT, D, VD);
}
